// FusedLogicTreeBlock_86277303042372
// MI455X (gfx1250) — compile-verified, matrix operand order checked by reading, not run on hardware
//
#include <hip/hip_runtime.h>
#include <hip/hip_bf16.h>
#include <stdint.h>

// ---------------- types ----------------
typedef __attribute__((ext_vector_type(16))) _Float16 v16h;
typedef __attribute__((ext_vector_type(8)))  float    v8f;
typedef __attribute__((ext_vector_type(4)))  uint32_t u32x4;
typedef __attribute__((ext_vector_type(8)))  int      i32x8;
typedef __attribute__((ext_vector_type(4)))  int      i32x4;

#define OC 128
#define CIN 64
#define HW 64
#define NLEAF 16
#define NGATE 15
#define ROWS 8              // y-rows per block
#define TPB 128             // 4 waves (wave32)

// op -> [c0, ca, cb, cab]
__device__ const float OP2POLY[16][4] = {
    {0,0,0,0},{0,0,0,1},{0,1,0,-1},{0,1,0,0},{0,0,1,-1},{0,0,1,0},{0,1,1,-2},{0,1,1,-1},
    {1,-1,-1,1},{1,-1,-1,2},{1,0,-1,0},{1,0,-1,1},{1,-1,0,0},{1,-1,0,1},{1,0,0,-1},{1,0,0,0}};

// =====================================================================
// Kernel 1: coeffs[og][0..3] = onehot(argmax(weights[og])) @ OP2POLY
// via V_WMMA_F32_16X16X32_F16 (exact: one-hot x small ints in f16).
// grid = 1920/16 = 120 blocks, 32 threads (one wave).
// =====================================================================
__global__ __launch_bounds__(32) void coeff_prep_wmma(
    const float* __restrict__ w, float* __restrict__ coeffs)
{
    const int lane = threadIdx.x;          // 0..31
    const int m    = lane & 15;            // matrix row / B column
    const bool hi  = lane >= 16;
    const int og   = blockIdx.x * 16 + m;  // flat (o*15+g), exactly 1920 rows

    // argmax over 16 weights (first-max tie-break like jnp.argmax)
    const float* row = w + og * 16;
    int kstar = 0; float best = row[0];
    #pragma unroll
    for (int k = 1; k < 16; ++k) { float v = row[k]; if (v > best) { best = v; kstar = k; } }

    // A: 16x32 f16 one-hot (K>=16 zero-padded); B: 32x16 f16 = OP2POLY padded.
    // 16-bit A layout: lane<16 -> elems 0..7 = K 0..7, 8..15 = K 16..23;
    //                  lane>=16 -> elems 0..7 = K 8..15, 8..15 = K 24..31.
    v16h a{}, bm{};
    #pragma unroll
    for (int i = 0; i < 16; ++i) {
        int K = (i < 8 ? i : i + 8) + (hi ? 8 : 0);
        a[i]  = (K == kstar) ? (_Float16)1.0f : (_Float16)0.0f;
        float pv = (K < 16 && m < 4) ? OP2POLY[K][m] : 0.0f;
        bm[i] = (_Float16)pv;
    }
    v8f c{};  // zero accumulator
    c = __builtin_amdgcn_wmma_f32_16x16x32_f16(false, a, false, bm,
                                               (short)0, c, false, false);
    // C/D f32 layout: VGPR r, lanes 0-15 -> (M=r, N=lane); lanes 16-31 -> (M=r+8)
    #pragma unroll
    for (int r = 0; r < 8; ++r) {
        int mm = r + (hi ? 8 : 0);
        if (m < 4) coeffs[(blockIdx.x * 16 + mm) * 4 + m] = c[r];
    }
}

// =====================================================================
// Kernel 2: gather->tree->pool. One block = (b, o, 8-row y stripe).
// LDS staging of the 16 leaf planes via TDM (iterate mode gives the
// 64-float LDS row stride; pre-zeroed LDS supplies the padding halo).
// =====================================================================
__global__ __launch_bounds__(TPB) void logic_tree_main(
    const float* __restrict__ x, const int* __restrict__ ci,
    const int* __restrict__ px, const int* __restrict__ py,
    const float* __restrict__ coeffs, float* __restrict__ out)
{
    __shared__ __align__(16) float sh[NLEAF * ROWS * 64];   // 32 KB leaf tiles
    __shared__ __align__(16) float scoef[NGATE * 4];

    const int t    = threadIdx.x;
    const int bid  = blockIdx.x;
    const int tile = bid & 7;
    const int o    = (bid >> 3) & (OC - 1);
    const int b    = bid >> 10;
    const int h0   = tile * ROWS;

    // zero LDS with b128 stores (provides the conv zero-padding halo)
    {
        float4* sh4 = (float4*)sh;
        const float4 z = make_float4(0.f, 0.f, 0.f, 0.f);
        #pragma unroll
        for (int i = 0; i < (NLEAF * ROWS * 64) / (4 * TPB); ++i)
            sh4[t + i * TPB] = z;
    }
    if (t < NGATE * 4) scoef[t] = coeffs[o * (NGATE * 4) + t];
    __syncthreads();

#if defined(__has_builtin) && __has_builtin(__builtin_amdgcn_tensor_load_to_lds)
    {
        // each of the 4 waves issues 4 TDM descriptors (TENSORcnt is per-wave)
        const int wv = t >> 5;
        const unsigned lds_base = (unsigned)(uintptr_t)(void*)sh; // LDS byte offset
        for (int n = wv * 4; n < wv * 4 + 4; ++n) {
            int c  = ci[o * NLEAF + n];
            int dx = px[o * NLEAF + n];
            int dy = py[o * NLEAF + n];
            int rstart = h0 + dy - 1;
            int r0 = rstart < 0 ? 0 : rstart;
            int r1 = rstart + ROWS; if (r1 > HW) r1 = HW;
            int nrows = r1 - r0;
            int cstart = dx - 1;
            int c0 = cstart < 0 ? 0 : cstart;
            int c1 = cstart + HW; if (c1 > HW) c1 = HW;
            int ncols = c1 - c0;
            unsigned long long gaddr = (unsigned long long)(uintptr_t)
                (x + (((size_t)b * CIN + c) * HW + r0) * HW + c0);
            unsigned lds_addr = lds_base +
                (unsigned)((n * ROWS * 64 + (r0 - rstart) * 64 + (c0 - cstart)) * 4);

            // ----- D# group 0: count=1 | lds_addr | global_addr | type=2 -----
            u32x4 g0;
            g0[0] = 1u;
            g0[1] = lds_addr;
            g0[2] = (uint32_t)gaddr;
            g0[3] = (uint32_t)((gaddr >> 32) & 0x1FFFFFFull) | (2u << 30);
            // ----- D# group 1: data_size=4B, iterate_enable, 1 x ncols tile -----
            i32x8 g1;
            g1[0] = (int)((2u << 16) | (1u << 19));      // data_size=2(4B), iterate=1
            g1[1] = 0;                                   // barrier addr 0, td0[15:0]=0
            g1[2] = 0x4000;                              // tensor_dim0 = 1<<30 (no OOB)
            g1[3] = (int)(0x4000u | ((unsigned)ncols << 16)); // td1=1<<30, tile_dim0=ncols
            g1[4] = 1;                                   // tile_dim1 = 1 row
            g1[5] = HW;                                  // tensor_dim0_stride = 64
            g1[6] = 0;
            g1[7] = 0;
            // ----- D# group 2 (iterate): lds_inc=64, glb_inc=64, count=nrows -----
            i32x4 g2;
            g2[0] = 0;
            g2[1] = 64;                                  // LDS row stride (elements)
            g2[2] = HW;                                  // global row stride (elements)
            g2[3] = (nrows - 1) << 16;                   // iterate_count (0 => 1x)
            i32x4 g3; g3[0] = 0; g3[1] = 0; g3[2] = 0; g3[3] = 0;
            // 6-arg toolchain form: extra (unused / VADDR4) group, zero-filled
            i32x8 g4;
            g4[0] = 0; g4[1] = 0; g4[2] = 0; g4[3] = 0;
            g4[4] = 0; g4[5] = 0; g4[6] = 0; g4[7] = 0;

            __builtin_amdgcn_tensor_load_to_lds(g0, g1, g2, g3, g4, 0);
        }
        __builtin_amdgcn_s_wait_tensorcnt(0);
    }
#else
    // Fallback: cooperative global->LDS gather with explicit bounds.
    {
        __shared__ int sc[NLEAF], sdx[NLEAF], sdy[NLEAF];
        if (t < NLEAF) {
            sc[t]  = ci[o * NLEAF + t];
            sdx[t] = px[o * NLEAF + t];
            sdy[t] = py[o * NLEAF + t];
        }
        __syncthreads();
        for (int i = t; i < NLEAF * ROWS * 64; i += TPB) {
            int n  = i >> 9;
            int rl = (i >> 6) & (ROWS - 1);
            int cl = i & 63;
            int r  = h0 + sdy[n] - 1 + rl;
            int cc = sdx[n] - 1 + cl;
            float v = 0.0f;
            if (r >= 0 && r < HW && cc >= 0 && cc < HW)
                v = x[(((size_t)b * CIN + sc[n]) * HW + r) * HW + cc];
            sh[i] = v;
        }
    }
#endif
    __syncthreads();

    // ---- compute: one pooled output per thread; 2x2 y-patch in registers ----
    const int pr = t >> 5;            // 0..3 pooled rows
    const int pc = t & 31;            // 0..31 pooled cols
    const int rb = pr * 2, cb = pc * 2;

    float v[NLEAF][4];
    #pragma unroll
    for (int n = 0; n < NLEAF; ++n) {
        #pragma unroll
        for (int dr = 0; dr < 2; ++dr) {
            // contiguous 8B-aligned pair -> ds_load_b64
            float2 p2 = *(const float2*)&sh[n * (ROWS * 64) + (rb + dr) * 64 + cb];
            v[n][dr * 2 + 0] = p2.x;
            v[n][dr * 2 + 1] = p2.y;
        }
    }

    int base = 0;
    #pragma unroll
    for (int lev = 0; lev < 4; ++lev) {
        const int ngl = 8 >> lev;
        #pragma unroll
        for (int i = 0; i < ngl; ++i) {
            // one ds_load_b128 broadcast per gate
            float4 cc4 = *(const float4*)&scoef[(base + i) * 4];
            #pragma unroll
            for (int p = 0; p < 4; ++p) {
                float av = v[2 * i][p], bv = v[2 * i + 1][p];
                v[i][p] = fmaf(cc4.w, av * bv, fmaf(cc4.z, bv, fmaf(cc4.y, av, cc4.x)));
            }
        }
        base += ngl;
    }

    float r = (1.0f - v[0][0]) * (1.0f - v[0][1]) *
              (1.0f - v[0][2]) * (1.0f - v[0][3]);
    out[(((size_t)b * OC + o) * 32 + (tile * 4 + pr)) * 32 + pc] = 1.0f - r;
}

// =====================================================================
extern "C" void kernel_launch(void* const* d_in, const int* in_sizes, int n_in,
                              void* d_out, int out_size, void* d_ws, size_t ws_size,
                              hipStream_t stream) {
    (void)in_sizes; (void)n_in; (void)out_size; (void)ws_size;
    const float* x  = (const float*)d_in[0];
    const float* w  = (const float*)d_in[1];
    const int*   ci = (const int*)d_in[2];
    const int*   px = (const int*)d_in[3];
    const int*   py = (const int*)d_in[4];
    float* out    = (float*)d_out;
    float* coeffs = (float*)d_ws;          // 1920*4 floats = 30 KB scratch

    coeff_prep_wmma<<<(OC * NGATE) / 16, 32, 0, stream>>>(w, coeffs);
    logic_tree_main<<<16 * OC * (HW / ROWS), TPB, 0, stream>>>(x, ci, px, py, coeffs, out);
}
